// neural_translation_model_67499706024619
// MI455X (gfx1250) — matrix operand order repaired and checked by reading, not compiled, not run on hardware
//
#include <hip/hip_runtime.h>
#include <hip/hip_bf16.h>
#include <math.h>

typedef _Float16 half_t;
typedef __attribute__((ext_vector_type(4)))  _Float16 v4h;
typedef __attribute__((ext_vector_type(8)))  _Float16 v8h;
typedef __attribute__((ext_vector_type(16))) _Float16 v16h;
typedef __attribute__((ext_vector_type(8)))  float    v8f;

constexpr int BB  = 64;    // batch
constexpr int TX  = 256;   // encoder time steps
constexpr int TY  = 64;    // decoder time steps
constexpr int NA  = 512;   // encoder units per direction
constexpr int NS  = 1024;  // decoder units
constexpr int HVd = 512;   // input feature width
constexpr int MVd = 1024;  // output vocab

__device__ __forceinline__ float sigf(float x){ return 1.0f/(1.0f+__expf(-x)); }

// =======================================================================
// Device-wide barrier (sense/generation) for persistent phase-stepped
// kernels. Every thread fences its global stores to device scope before
// arrival; agent-scope atomics; s_sleep backoff while spinning. All blocks
// are resident (<=64 small blocks on an MI455X-class part) so no deadlock.
// =======================================================================
__device__ __forceinline__ void gbar(unsigned* cnt, unsigned* gen, unsigned nblocks){
  __threadfence();                  // all threads: stores visible device-wide
  __syncthreads();
  if (threadIdx.x == 0){
    unsigned g = __hip_atomic_load(gen, __ATOMIC_RELAXED, __HIP_MEMORY_SCOPE_AGENT);
    unsigned a = __hip_atomic_fetch_add(cnt, 1u, __ATOMIC_ACQ_REL, __HIP_MEMORY_SCOPE_AGENT);
    if (a == nblocks - 1u){
      __hip_atomic_store(cnt, 0u, __ATOMIC_RELAXED, __HIP_MEMORY_SCOPE_AGENT);
      __hip_atomic_fetch_add(gen, 1u, __ATOMIC_ACQ_REL, __HIP_MEMORY_SCOPE_AGENT);
    } else {
      while (__hip_atomic_load(gen, __ATOMIC_ACQUIRE, __HIP_MEMORY_SCOPE_AGENT) == g){
        __builtin_amdgcn_s_sleep(2);
      }
    }
    __threadfence();
  }
  __syncthreads();
}

// =======================================================================
// WMMA strip GEMM (device function): one wave computes a 16x64 strip of
// C[MxN] = [A1 | A2] x Bsw + bias. A fragment = two contiguous b128 loads
// from row-major f16; B fragment = one contiguous 32B lane load from the
// pre-swizzled weights; 4 v_wmma_f32_16x16x32_f16 per K-tile (A reuse x4).
// The entry memory clobber stops the compiler from hoisting the (const,
// restrict) B-fragment loads out of the persistent kernels' time-step loop
// -- without it, clang preloads all Kt fragments and spills them to
// scratch, turning L2-resident global_load_b128 into scratch reloads.
// Caller guarantees whole-wave participation (EXEC all ones).
// =======================================================================
__device__ __forceinline__ void gemm_strip(int wave, int lane,
                            const half_t* __restrict__ A1, long ld1, int K1,
                            const half_t* __restrict__ A2, long ld2, int K2,
                            const half_t* __restrict__ Bsw,
                            const float* __restrict__ bias,
                            float* __restrict__ C, int ldc, int M, int N){
  asm volatile("" ::: "memory");         // defeat cross-step load hoist/spill
  int mTiles = M >> 4;
  int mt = wave % mTiles;
  int ng = wave / mTiles;
  int m     = (mt << 4) + (lane & 15);
  int khOff = (lane >> 4) << 4;          // 0 or 16 halves (A lane-half K offset)
  int K  = K1 + K2;
  int Kt = K >> 5;
  size_t bcol = (size_t)Kt * 512;        // halves per B column-tile
  const half_t* bp = Bsw + (size_t)(ng*4)*bcol + (size_t)lane*16;
  v8f acc0 = {}, acc1 = {}, acc2 = {}, acc3 = {};
  for (int kt = 0; kt < Kt; ++kt){
    int k0 = kt << 5;
    const half_t* ap = (k0 < K1) ? (A1 + (size_t)m*ld1 + k0)
                                 : (A2 + (size_t)m*ld2 + (k0 - K1));
    v8h lo = *(const v8h*)(ap + khOff);       // K = kh*16 .. +7     (VGPR 0..3)
    v8h hi = *(const v8h*)(ap + 8 + khOff);   // K = kh*16 + 8 .. +7 (VGPR 4..7)
    v16h a  = __builtin_shufflevector(lo, hi, 0,1,2,3,4,5,6,7,8,9,10,11,12,13,14,15);
    v16h b0 = *(const v16h*)(bp);
    v16h b1 = *(const v16h*)(bp +   bcol);
    v16h b2 = *(const v16h*)(bp + 2*bcol);
    v16h b3 = *(const v16h*)(bp + 3*bcol);
    if (kt + 1 < Kt) __builtin_prefetch(bp + 512, 0, 1);
    acc0 = __builtin_amdgcn_wmma_f32_16x16x32_f16(false, a, false, b0, (short)0, acc0, false, false);
    acc1 = __builtin_amdgcn_wmma_f32_16x16x32_f16(false, a, false, b1, (short)0, acc1, false, false);
    acc2 = __builtin_amdgcn_wmma_f32_16x16x32_f16(false, a, false, b2, (short)0, acc2, false, false);
    acc3 = __builtin_amdgcn_wmma_f32_16x16x32_f16(false, a, false, b3, (short)0, acc3, false, false);
    bp += 512;
  }
  int n0 = ng << 6;
  int nl = lane & 15;
  int mh = (lane >> 4) << 3;
  float bv0 = bias[n0+nl], bv1 = bias[n0+16+nl], bv2 = bias[n0+32+nl], bv3 = bias[n0+48+nl];
  #pragma unroll
  for (int r = 0; r < 8; ++r){
    float* crow = C + (size_t)((mt<<4) + mh + r)*ldc + n0;
    crow[nl]      = acc0[r] + bv0;
    crow[16 + nl] = acc1[r] + bv1;
    crow[32 + nl] = acc2[r] + bv2;
    crow[48 + nl] = acc3[r] + bv3;
  }
}

// ---------------- elementwise f32 -> f16 ----------------
__global__ void k_cvt_f16(const float* __restrict__ in, half_t* __restrict__ out, size_t n){
  size_t i = (size_t)blockIdx.x*blockDim.x + threadIdx.x;
  if (i < n) out[i] = (half_t)in[i];
}

// ---------------- weight swizzle into WMMA B-fragment tile order ----------------
// B-matrix of V_WMMA_F32_16X16X32_F16 is 32x16 (KxN): here lane L holds the 16
// consecutive K values of its lane-half for column n=(L&15) (contiguous, as the A loads):
//   k = kt*32 + (L>>4)*16 + j ,  n = nt*16 + (L&15)
// stored so each lane's fragment is 32 contiguous bytes. Two stacked sources
// ([Wx ; Wh]) fold x@Wx + h@Wh into one GEMM.
__global__ void k_swz_b(const float* __restrict__ Wa, int Ka,
                        const float* __restrict__ Wb, int Kb,
                        half_t* __restrict__ out, int N){
  size_t idx = (size_t)blockIdx.x*blockDim.x + threadIdx.x;
  int K = Ka + Kb;
  size_t total = (size_t)K * N;
  if (idx >= total) return;
  int j    = (int)(idx & 15);
  int lane = (int)((idx >> 4) & 31);
  size_t tile = idx >> 9;
  int Kt = K >> 5;
  int nt = (int)(tile / Kt);
  int kt = (int)(tile % Kt);
  int k = kt*32 + ((lane>>4)<<4) + j;
  int n = nt*16 + (lane & 15);
  float v = (k < Ka) ? Wa[(size_t)k*N + n] : Wb[(size_t)(k-Ka)*N + n];
  out[idx] = (half_t)v;
}

// ---------------- attention: hoisted a @ W1a (a is fixed over all Ty) ----------------
__global__ void k_attn_aW1(const half_t* __restrict__ a_buf,
                           const float* __restrict__ W1,   // (2048 x 10)
                           float* __restrict__ aW1){
  int idx = blockIdx.x*blockDim.x + threadIdx.x;            // BB*TX*10
  if (idx >= BB*TX*10) return;
  int j = idx % 10, row = idx / 10;
  const half_t* ar = a_buf + (size_t)row*(2*NA);
  float acc = 0.f;
  for (int k = 0; k < 2*NA; ++k) acc += (float)ar[k] * W1[(size_t)k*10 + j];
  aW1[idx] = acc;
}

// ---------------- deterministic state + barrier init (every launch) ----------------
__global__ void k_init(const float* __restrict__ s0, const float* __restrict__ c0,
                       float* __restrict__ s, float* __restrict__ cdec,
                       half_t* __restrict__ s_half, half_t* __restrict__ Adec,
                       float* __restrict__ cf, float* __restrict__ cb,
                       half_t* __restrict__ hf_half, half_t* __restrict__ hb_half,
                       unsigned* __restrict__ bar){
  int idx = blockIdx.x*blockDim.x + threadIdx.x;            // BB*NS threads
  if (idx == 0){ bar[0] = 0u; bar[1] = 0u; }
  if (idx < BB*NS){
    float v = s0[idx];
    s[idx] = v; cdec[idx] = c0[idx]; s_half[idx] = (half_t)v;
    int b = idx >> 10, u = idx & (NS-1);
    Adec[(size_t)b*(2*NS) + NS + u] = (half_t)v;
  }
  if (idx < BB*NA){
    cf[idx] = 0.f; cb[idx] = 0.f;
    hf_half[idx] = (half_t)0.f; hb_half[idx] = (half_t)0.f;
  }
}

// =======================================================================
// Persistent bidirectional encoder: ONE launch runs all 256 time steps.
// Grid: 64 blocks x 128 threads. Blocks 0..31 forward, 32..63 backward.
// Per step: (a) 128 waves/dir compute the 64x2048 stacked-K gate GEMM with
// WMMA, (b) device barrier, (c) fused LSTM pointwise emits c (f32) and h
// (f16, both next-step A segment and attention buffer), (d) device barrier.
// =======================================================================
__global__ void k_encoder_persist(const half_t* __restrict__ Xh,
                                  const half_t* __restrict__ Wenc_f,
                                  const half_t* __restrict__ Wenc_b,
                                  const float* __restrict__ b_f,
                                  const float* __restrict__ b_b,
                                  float* __restrict__ gates_f,
                                  float* __restrict__ gates_b,
                                  float* __restrict__ cf, float* __restrict__ cb,
                                  half_t* __restrict__ hf_half,
                                  half_t* __restrict__ hb_half,
                                  half_t* __restrict__ a_buf,
                                  unsigned* __restrict__ bar){
  const int dir  = (blockIdx.x >= 32) ? 1 : 0;
  const int blk  = blockIdx.x & 31;
  const int wave = blk*4 + ((int)threadIdx.x >> 5);   // 0..127 within direction
  const int lane = (int)threadIdx.x & 31;
  const half_t* Wsw   = dir ? Wenc_b  : Wenc_f;
  const float*  bias  = dir ? b_b     : b_f;
  float*        gates = dir ? gates_b : gates_f;
  float*        cst   = dir ? cb      : cf;
  half_t*       hh16  = dir ? hb_half : hf_half;
  const long ldX = (long)TX*HVd;

  for (int step = 0; step < TX; ++step){
    int t = dir ? (TX-1-step) : step;
    // ---- gate GEMM: 64 x 2048, K = 512(x_t) + 512(h_{t-1}) ----
    gemm_strip(wave, lane,
               Xh + (size_t)t*HVd, ldX, HVd,
               hh16, NA, NA,
               Wsw, bias, gates, 4*NA, BB, 4*NA);
    gbar(bar, bar+1, 64);
    // ---- fused LSTM pointwise: 32768 elems over 4096 threads/dir ----
    #pragma unroll
    for (int i = 0; i < 8; ++i){
      int idx = i*4096 + blk*128 + (int)threadIdx.x;   // 0..32767
      int b = idx >> 9, u = idx & (NA-1);
      const float* g = gates + (size_t)b*(4*NA);
      float ig = sigf(g[u]);
      float fg = sigf(g[NA + u]);
      float gg = tanhf(g[2*NA + u]);
      float og = sigf(g[3*NA + u]);
      float cc = fg*cst[idx] + ig*gg;
      float hv = og*tanhf(cc);
      cst[idx] = cc;
      hh16[idx] = (half_t)hv;
      a_buf[((size_t)b*TX + t)*(2*NA) + dir*NA + u] = (half_t)hv;
    }
    gbar(bar, bar+1, 64);
  }
}

// =======================================================================
// Persistent decoder: ONE launch runs all 64 decoder steps, 7 phases each.
// Grid: 64 blocks x 256 threads (512 waves). Block b doubles as the batch
// row owner for the block-local softmax phases.
// =======================================================================
__global__ void k_decoder_persist(const half_t* __restrict__ a_buf,
                                  const float* __restrict__ aW1,
                                  const float* __restrict__ W1,
                                  const float* __restrict__ b1,
                                  const float* __restrict__ W2,
                                  const float* __restrict__ b2,
                                  const half_t* __restrict__ Wdec,
                                  const float* __restrict__ b_d,
                                  const half_t* __restrict__ Wo_sw,
                                  const float* __restrict__ bo,
                                  float* __restrict__ s_st,
                                  float* __restrict__ c_dec,
                                  half_t* __restrict__ s_half,
                                  half_t* __restrict__ Adec,
                                  float* __restrict__ gates_d,
                                  float* __restrict__ logits,
                                  float* __restrict__ sW1,
                                  float* __restrict__ energ,
                                  float* __restrict__ out,
                                  unsigned* __restrict__ bar){
  const int tid = (int)threadIdx.x;
  const int b   = (int)blockIdx.x;
  const int gthread = b*256 + tid;      // 0..16383
  const int wave    = gthread >> 5;     // 0..511
  const int lane    = tid & 31;
  __shared__ float sh[256];
  __shared__ float alpha[256];

  for (int ty = 0; ty < TY; ++ty){
    // ---- Phase 1: sW1 = s @ W1s (rows 1024..2047 of W1), 640 dots ----
    if (gthread < BB*10){
      int j = gthread % 10, bb = gthread / 10;
      const float* sr = s_st + (size_t)bb*NS;
      float acc = 0.f;
      for (int k = 0; k < NS; ++k) acc += sr[k] * W1[(size_t)(2*NA + k)*10 + j];
      sW1[gthread] = acc;
    }
    gbar(bar, bar+1, 64);
    // ---- Phase 2: energies = relu(tanh(aW1+sW1+b1) @ W2 + b2), 16384 ----
    {
      int bb = gthread >> 8;
      float en = b2[0];
      #pragma unroll
      for (int j = 0; j < 10; ++j){
        float e = tanhf(aW1[(size_t)gthread*10 + j] + sW1[bb*10 + j] + b1[j]);
        en += e * W2[j];
      }
      energ[gthread] = fmaxf(en, 0.f);
    }
    gbar(bar, bar+1, 64);
    // ---- Phase 3: softmax over Tx + context = alpha . a (block b = row b) ----
    {
      float e = energ[b*TX + tid];
      sh[tid] = e; __syncthreads();
      for (int s = 128; s > 0; s >>= 1){ if (tid < s) sh[tid] = fmaxf(sh[tid], sh[tid+s]); __syncthreads(); }
      float mx = sh[0]; __syncthreads();
      float ex = __expf(e - mx);
      sh[tid] = ex; __syncthreads();
      for (int s = 128; s > 0; s >>= 1){ if (tid < s) sh[tid] += sh[tid+s]; __syncthreads(); }
      float inv = 1.0f / sh[0];
      alpha[tid] = ex * inv; __syncthreads();
      // context: thread owns 4 contiguous d -> one v4h (8B) load per t
      const half_t* arow = a_buf + (size_t)b*TX*(2*NA);
      int d0 = tid * 4;                                 // 256 threads x 4 = 1024
      float a0 = 0.f, a1 = 0.f, a2 = 0.f, a3 = 0.f;
      for (int t = 0; t < TX; ++t){
        v4h av = *(const v4h*)(arow + (size_t)t*(2*NA) + d0);
        float al = alpha[t];
        a0 += al * (float)av[0];
        a1 += al * (float)av[1];
        a2 += al * (float)av[2];
        a3 += al * (float)av[3];
      }
      v4h cv; cv[0] = (half_t)a0; cv[1] = (half_t)a1; cv[2] = (half_t)a2; cv[3] = (half_t)a3;
      *(v4h*)(Adec + (size_t)b*(2*NS) + d0) = cv;       // context half of [context | s]
    }
    gbar(bar, bar+1, 64);
    // ---- Phase 4: gate GEMM 64 x 4096, K = 2048 ([context|s]) : 256 waves ----
    if (wave < 256){
      gemm_strip(wave, lane,
                 Adec, 2*NS, 2*NS,
                 Adec, 2*NS, 0,
                 Wdec, b_d, gates_d, 4*NS, BB, 4*NS);
    }
    gbar(bar, bar+1, 64);
    // ---- Phase 5: fused LSTM pointwise: 65536 elems / 16384 threads ----
    #pragma unroll
    for (int i = 0; i < 4; ++i){
      int idx = i*16384 + gthread;     // 0..65535
      int bb = idx >> 10, u = idx & (NS-1);
      const float* g = gates_d + (size_t)bb*(4*NS);
      float ig = sigf(g[u]);
      float fg = sigf(g[NS + u]);
      float gg = tanhf(g[2*NS + u]);
      float og = sigf(g[3*NS + u]);
      float cc = fg*c_dec[idx] + ig*gg;
      float hv = og*tanhf(cc);
      c_dec[idx] = cc;
      s_st[idx]  = hv;
      s_half[idx] = (half_t)hv;
      Adec[(size_t)bb*(2*NS) + NS + u] = (half_t)hv;
    }
    gbar(bar, bar+1, 64);
    // ---- Phase 6: logits = s @ Wo + bo (64 x 1024, K=1024) : 64 waves ----
    if (wave < 64){
      gemm_strip(wave, lane,
                 s_half, NS, NS,
                 s_half, NS, 0,
                 Wo_sw, bo, logits, MVd, BB, MVd);
    }
    gbar(bar, bar+1, 64);
    // ---- Phase 7: output softmax over MV, write time slice (block b = row b) ----
    {
      const float* row = logits + (size_t)b*MVd;
      float mx = -INFINITY;
      #pragma unroll
      for (int i = 0; i < 4; ++i) mx = fmaxf(mx, row[tid + i*256]);
      sh[tid] = mx; __syncthreads();
      for (int s = 128; s > 0; s >>= 1){ if (tid < s) sh[tid] = fmaxf(sh[tid], sh[tid+s]); __syncthreads(); }
      mx = sh[0]; __syncthreads();
      float ex[4]; float sum = 0.f;
      #pragma unroll
      for (int i = 0; i < 4; ++i){ ex[i] = __expf(row[tid + i*256] - mx); sum += ex[i]; }
      sh[tid] = sum; __syncthreads();
      for (int s = 128; s > 0; s >>= 1){ if (tid < s) sh[tid] += sh[tid+s]; __syncthreads(); }
      float inv = 1.0f / sh[0];
      float* orow = out + ((size_t)b*TY + ty)*MVd;
      #pragma unroll
      for (int i = 0; i < 4; ++i) orow[tid + i*256] = ex[i] * inv;
    }
    gbar(bar, bar+1, 64);
  }
}

extern "C" void kernel_launch(void* const* d_in, const int* in_sizes, int n_in,
                              void* d_out, int out_size, void* d_ws, size_t ws_size,
                              hipStream_t stream){
  (void)in_sizes; (void)n_in; (void)out_size; (void)ws_size;
  const float* X    = (const float*)d_in[0];
  const float* s0   = (const float*)d_in[1];
  const float* c0   = (const float*)d_in[2];
  const float* Wx_f = (const float*)d_in[3];
  const float* Wh_f = (const float*)d_in[4];
  const float* b_f  = (const float*)d_in[5];
  const float* Wx_b = (const float*)d_in[6];
  const float* Wh_b = (const float*)d_in[7];
  const float* b_b  = (const float*)d_in[8];
  const float* W1   = (const float*)d_in[9];
  const float* b1   = (const float*)d_in[10];
  const float* W2   = (const float*)d_in[11];
  const float* b2   = (const float*)d_in[12];
  const float* Wx_d = (const float*)d_in[13];
  const float* Wh_d = (const float*)d_in[14];
  const float* b_d  = (const float*)d_in[15];
  const float* Wo   = (const float*)d_in[16];
  const float* bo   = (const float*)d_in[17];
  float* out = (float*)d_out;

  // ---- workspace carve-up (256B aligned) ----
  char* ws = (char*)d_ws;
  size_t off = 0;
  auto alloc = [&](size_t bytes)->void*{
    void* p = ws + off; off = (off + bytes + 255) & ~(size_t)255; return p;
  };
  half_t* Xh      = (half_t*)alloc((size_t)BB*TX*HVd*2);
  half_t* Wenc_f  = (half_t*)alloc((size_t)(HVd+NA)*(4*NA)*2);
  half_t* Wenc_b  = (half_t*)alloc((size_t)(HVd+NA)*(4*NA)*2);
  half_t* Wdec    = (half_t*)alloc((size_t)(2*NA+NS)*(4*NS)*2);
  half_t* Wo_sw   = (half_t*)alloc((size_t)NS*MVd*2);
  half_t* a_buf   = (half_t*)alloc((size_t)BB*TX*(2*NA)*2);
  half_t* hf_half = (half_t*)alloc((size_t)BB*NA*2);
  half_t* hb_half = (half_t*)alloc((size_t)BB*NA*2);
  half_t* s_half  = (half_t*)alloc((size_t)BB*NS*2);
  half_t* Adec    = (half_t*)alloc((size_t)BB*(2*NS)*2);
  float*  cf      = (float*)alloc((size_t)BB*NA*4);
  float*  cb      = (float*)alloc((size_t)BB*NA*4);
  float*  s_st    = (float*)alloc((size_t)BB*NS*4);
  float*  c_dec   = (float*)alloc((size_t)BB*NS*4);
  float*  gates_f = (float*)alloc((size_t)BB*(4*NA)*4);
  float*  gates_b = (float*)alloc((size_t)BB*(4*NA)*4);
  float*  gates_d = (float*)alloc((size_t)BB*(4*NS)*4);
  float*  logits  = (float*)alloc((size_t)BB*MVd*4);
  float*  aW1     = (float*)alloc((size_t)BB*TX*10*4);
  float*  sW1     = (float*)alloc((size_t)BB*10*4);
  float*  energ   = (float*)alloc((size_t)BB*TX*4);
  unsigned* bar   = (unsigned*)alloc(256);

  // ---- prep: conversions, weight swizzles, state/barrier init ----
  {
    size_t n = (size_t)BB*TX*HVd;
    k_cvt_f16<<<(int)((n+255)/256), 256, 0, stream>>>(X, Xh, n);
  }
  k_swz_b<<<(int)(((size_t)(HVd+NA)*(4*NA)+255)/256), 256, 0, stream>>>(Wx_f, HVd, Wh_f, NA, Wenc_f, 4*NA);
  k_swz_b<<<(int)(((size_t)(HVd+NA)*(4*NA)+255)/256), 256, 0, stream>>>(Wx_b, HVd, Wh_b, NA, Wenc_b, 4*NA);
  k_swz_b<<<(int)(((size_t)(2*NA+NS)*(4*NS)+255)/256), 256, 0, stream>>>(Wx_d, 2*NA, Wh_d, NS, Wdec, 4*NS);
  k_swz_b<<<(int)(((size_t)NS*MVd+255)/256), 256, 0, stream>>>(Wo, NS, Wo, 0, Wo_sw, MVd);
  k_init<<<(BB*NS+255)/256, 256, 0, stream>>>(s0, c0, s_st, c_dec, s_half, Adec, cf, cb, hf_half, hb_half, bar);

  // ---- persistent encoder: 256 steps, both directions, one launch ----
  k_encoder_persist<<<64, 128, 0, stream>>>(Xh, Wenc_f, Wenc_b, b_f, b_b,
                                            gates_f, gates_b, cf, cb,
                                            hf_half, hb_half, a_buf, bar);

  // ---- hoist a@W1a out of the decoder loop ----
  k_attn_aW1<<<(BB*TX*10+255)/256, 256, 0, stream>>>(a_buf, W1, aW1);

  // ---- persistent decoder: 64 steps x 7 phases, one launch ----
  k_decoder_persist<<<64, 256, 0, stream>>>(a_buf, aW1, W1, b1, W2, b2,
                                            Wdec, b_d, Wo_sw, bo,
                                            s_st, c_dec, s_half, Adec,
                                            gates_d, logits, sW1, energ,
                                            out, bar);
}
